// MSA_50508815401350
// MI455X (gfx1250) — matrix operand order repaired and checked by reading, not compiled, not run on hardware
//
#include <hip/hip_runtime.h>
#include <math.h>

typedef _Float16 half_t;
typedef __attribute__((ext_vector_type(16))) _Float16 v16h;
typedef __attribute__((ext_vector_type(8)))  _Float16 v8h;
typedef __attribute__((ext_vector_type(8)))  float    v8f;
typedef __attribute__((ext_vector_type(4)))  int      v4i;

// ---------------------------------------------------------------------------
// WMMA fragment helpers (CDNA5 16x16x32 f16, wave32)
// A (16x32, MxK): lane L -> row M = L%16, kb = (L/16)*8; elems 0..7 = K kb..kb+7,
//                 elems 8..15 = K kb+16..kb+23.
// B (32x16, KxN): lane L -> col N = L%16, kb = (L/16)*8; same K split as A.
// C/D (16x16):    lane L -> col N = L%16, elem v -> row M = v + 8*(L/16).
// ---------------------------------------------------------------------------

__device__ __forceinline__ v16h load_fragA(const half_t* __restrict__ base, int ld) {
  const int lane = threadIdx.x & 31;
  const int r    = lane & 15;
  const int kb   = (lane >> 4) << 3;          // 0 or 8
  const half_t* p = base + (size_t)r * ld + kb;
  v8h lo = *(const v8h*)(p);                   // K = kb .. kb+7
  v8h hi = *(const v8h*)(p + 16);              // K = kb+16 .. kb+23
  return __builtin_shufflevector(lo, hi, 0,1,2,3,4,5,6,7,8,9,10,11,12,13,14,15);
}

__device__ __forceinline__ v16h load_fragB(const half_t* __restrict__ base, int ld) {
  const int lane = threadIdx.x & 31;
  const int r    = lane & 15;
  const int kb   = (lane >> 4) << 3;          // 0 or 8
  const half_t* p = base + (size_t)r * ld + kb;
  v8h lo = *(const v8h*)(p);                   // K = kb .. kb+7
  v8h hi = *(const v8h*)(p + 16);              // K = kb+16 .. kb+23
  return __builtin_shufflevector(lo, hi, 0,1,2,3,4,5,6,7,8,9,10,11,12,13,14,15);
}

__device__ __forceinline__ v8f wmma16(v16h a, v16h b, v8f c) {
  return __builtin_amdgcn_wmma_f32_16x16x32_f16(false, a, false, b, (short)0, c, false, false);
}

__device__ __forceinline__ float rmax16(float x) {
#pragma unroll
  for (int m = 1; m < 16; m <<= 1) x = fmaxf(x, __shfl_xor(x, m, 32));
  return x;
}
__device__ __forceinline__ float rsum16(float x) {
#pragma unroll
  for (int m = 1; m < 16; m <<= 1) x += __shfl_xor(x, m, 32);
  return x;
}

// ---------------------------------------------------------------------------
// CDNA5 async global->LDS copy (16B per lane), tracked by ASYNCcnt.
// ---------------------------------------------------------------------------
typedef __attribute__((address_space(1))) v4i* gv4i_p;   // global
typedef __attribute__((address_space(3))) v4i* lv4i_p;   // LDS

__device__ __forceinline__ void async_b128(half_t* lds, const half_t* g) {
#if __has_builtin(__builtin_amdgcn_global_load_async_to_lds_b128)
  __builtin_amdgcn_global_load_async_to_lds_b128((gv4i_p)g, (lv4i_p)lds, 0, 0);
#else
  asm volatile("global_load_async_to_lds_b128 %0, %1, off"
               :: "v"((lv4i_p)lds), "v"((gv4i_p)g)
               : "memory");
#endif
}

__device__ __forceinline__ void wait_async0() {
#if __has_builtin(__builtin_amdgcn_s_wait_asynccnt)
  __builtin_amdgcn_s_wait_asynccnt(0);
#else
  asm volatile("s_wait_asynccnt 0" ::: "memory");
#endif
}

// ---------------------------------------------------------------------------
// fp32 -> f16 convert
// ---------------------------------------------------------------------------
__global__ void msa_cvt_f32_f16(const float* __restrict__ src,
                                half_t* __restrict__ dst, int n) {
  int i = blockIdx.x * blockDim.x + threadIdx.x;
  if (i < n) dst[i] = (half_t)src[i];
}

// ---------------------------------------------------------------------------
// QKV projection: Y[s,e] = sum_i x[s,i] * W[e,i]
// grid = (512 row tiles, 3 matrices), block = 256 (8 waves; wave w -> cols w*64..)
// Q,K stored [b,h,s,64] f16; V stored transposed [b,h,64,s] f16.
// ---------------------------------------------------------------------------
__global__ void __launch_bounds__(256)
msa_proj_qkv(const half_t* __restrict__ xh,
             const half_t* __restrict__ WQ, const half_t* __restrict__ WK,
             const half_t* __restrict__ WV,
             half_t* __restrict__ Qh, half_t* __restrict__ Kh,
             half_t* __restrict__ Vth) {
  const int wave   = threadIdx.x >> 5;
  const int lane   = threadIdx.x & 31;
  const int r      = lane & 15;
  const int halfid = lane >> 4;
  const int row0   = blockIdx.x * 16;       // global row in [0, 8192)
  const int mat    = blockIdx.y;            // 0=Q 1=K 2=V
  const int col0   = wave * 64;

  const half_t* W = (mat == 0) ? WQ : (mat == 1) ? WK : WV;

  v8f acc[4] = {v8f{}, v8f{}, v8f{}, v8f{}};
  for (int k = 0; k < 512; k += 32) {
    v16h a = load_fragA(xh + (size_t)row0 * 512 + k, 512);
#pragma unroll
    for (int j = 0; j < 4; ++j) {
      v16h bw = load_fragB(W + (size_t)(col0 + j * 16) * 512 + k, 512);
      acc[j] = wmma16(a, bw, acc[j]);
    }
  }

  const int b  = row0 >> 12;
  const int s0 = row0 & 4095;
#pragma unroll
  for (int j = 0; j < 4; ++j) {
    const int col = col0 + j * 16 + r;
    const int h = col >> 6, d = col & 63;
#pragma unroll
    for (int v = 0; v < 8; ++v) {
      const int s = s0 + v + 8 * halfid;
      const half_t val = (half_t)acc[j][v];
      if (mat == 2) {
        Vth[(((size_t)(b * 8 + h)) * 64 + d) * 4096 + s] = val;
      } else {
        half_t* dst = (mat == 0) ? Qh : Kh;
        dst[(((size_t)(b * 8 + h)) * 4096 + s) * 64 + d] = val;
      }
    }
  }
}

// ---------------------------------------------------------------------------
// Fused flash attention. grid = (32, H=8, B=2), block = 256 (8 waves).
// All 8 waves share one (b,h); K/V tiles staged once per block into LDS via
// async DMA (double-buffered), fragments read back with ds_load_b128.
// ---------------------------------------------------------------------------
__global__ void __launch_bounds__(256)
msa_attention(const half_t* __restrict__ Qh, const half_t* __restrict__ Kh,
              const half_t* __restrict__ Vth, half_t* __restrict__ wVh) {
  __shared__ __align__(16) half_t shK[2][32 * 64];   // keys x d       (4KB each)
  __shared__ __align__(16) half_t shV[2][64 * 32];   // d x keys       (4KB each)
  __shared__ __align__(16) half_t shP[8][16 * 32];   // per-wave P tile

  const int tid    = threadIdx.x;
  const int wave   = tid >> 5;
  const int lane   = tid & 31;
  const int r      = lane & 15;
  const int halfid = lane >> 4;
  const int h  = blockIdx.y;
  const int b  = blockIdx.z;
  const int q0 = (blockIdx.x * 8 + wave) * 16;

  const size_t headoff = ((size_t)(b * 8 + h)) * 4096 * 64;
  const half_t* Qp = Qh + headoff + (size_t)q0 * 64;
  const half_t* Kp = Kh + headoff;            // [4096, 64]
  const half_t* Vp = Vth + headoff;           // [64, 4096]
  half_t* myP = shP[wave];

  // cooperative stage of one 32-key tile (K: 32x64, V: 64x32) into buffer buf
  const int krow = tid >> 3, kc8 = (tid & 7) << 3;   // 32 rows x 8 cols of 8h
  const int vrow = tid >> 2, vc8 = (tid & 3) << 3;   // 64 rows x 4 cols of 8h
#define STAGE(buf, k0)                                                         \
  do {                                                                         \
    async_b128(&shK[buf][krow * 64 + kc8], Kp + (size_t)((k0) + krow) * 64 + kc8); \
    async_b128(&shV[buf][vrow * 32 + vc8], Vp + (size_t)vrow * 4096 + (k0) + vc8); \
  } while (0)

  STAGE(0, 0);

  v16h a0 = load_fragA(Qp + 0, 64);
  v16h a1 = load_fragA(Qp + 32, 64);

  v8f o0 = {}, o1 = {}, o2 = {}, o3 = {};
  float m[8], l[8];
#pragma unroll
  for (int v = 0; v < 8; ++v) { m[v] = -INFINITY; l[v] = 0.0f; }

  const float inv_e = 1.0f / 512.0f;          // reference scales by full E

  for (int it = 0; it < 128; ++it) {
    const int buf = it & 1;
    wait_async0();            // this wave's DMA into buf complete
    __syncthreads();          // everyone's DMA into buf complete; buf^1 reads done
    if (it + 1 < 128) STAGE(buf ^ 1, (it + 1) * 32);

    // ---- scores: two 16x16 tiles over 32 staged keys ----------------------
    v8f s0 = {}, s1 = {};
    {
      v16h bk;
      bk = load_fragB(&shK[buf][0 * 64 + 0],  64); s0 = wmma16(a0, bk, s0);
      bk = load_fragB(&shK[buf][0 * 64 + 32], 64); s0 = wmma16(a1, bk, s0);
      bk = load_fragB(&shK[buf][16 * 64 + 0],  64); s1 = wmma16(a0, bk, s1);
      bk = load_fragB(&shK[buf][16 * 64 + 32], 64); s1 = wmma16(a1, bk, s1);
    }
    // ---- online softmax ---------------------------------------------------
#pragma unroll
    for (int v = 0; v < 8; ++v) {
      const float x0 = s0[v] * inv_e;
      const float x1 = s1[v] * inv_e;
      float rm = rmax16(fmaxf(x0, x1));
      const float mn = fmaxf(m[v], rm);
      const float p0 = __expf(x0 - mn);
      const float p1 = __expf(x1 - mn);
      const float rs = rsum16(p0 + p1);
      const float alpha = __expf(m[v] - mn);
      l[v] = l[v] * alpha + rs;
      m[v] = mn;
      o0[v] *= alpha; o1[v] *= alpha; o2[v] *= alpha; o3[v] *= alpha;
      const int row = v + 8 * halfid;
      myP[row * 32 + r]      = (half_t)p0;
      myP[row * 32 + r + 16] = (half_t)p1;
    }
    asm volatile("s_wait_dscnt 0" ::: "memory");
    // ---- P (A layout via LDS) x V ----------------------------------------
    v16h pf = load_fragA(myP, 32);
    {
      v16h bv;
      bv = load_fragB(&shV[buf][0 * 32], 32);  o0 = wmma16(pf, bv, o0);
      bv = load_fragB(&shV[buf][16 * 32], 32); o1 = wmma16(pf, bv, o1);
      bv = load_fragB(&shV[buf][32 * 32], 32); o2 = wmma16(pf, bv, o2);
      bv = load_fragB(&shV[buf][48 * 32], 32); o3 = wmma16(pf, bv, o3);
    }
  }
#undef STAGE

  // ---- normalize and store wV[b*S + s, h*64 + d] as f16 -------------------
  const size_t outrow = (size_t)b * 4096 + q0;
#pragma unroll
  for (int v = 0; v < 8; ++v) {
    const float invl = 1.0f / l[v];
    const int row = v + 8 * halfid;
    half_t* dst = wVh + (outrow + row) * 512 + h * 64 + r;
    dst[0]  = (half_t)(o0[v] * invl);
    dst[16] = (half_t)(o1[v] * invl);
    dst[32] = (half_t)(o2[v] * invl);
    dst[48] = (half_t)(o3[v] * invl);
  }
}

// ---------------------------------------------------------------------------
// Output projection: out[s,o] = sum_e wV[s,e] * W_O[o,e], fp32 store.
// grid = 512 row tiles, block = 256 (8 waves; wave w -> cols w*64..)
// ---------------------------------------------------------------------------
__global__ void __launch_bounds__(256)
msa_proj_out(const half_t* __restrict__ wVh, const half_t* __restrict__ WO,
             float* __restrict__ out) {
  const int wave   = threadIdx.x >> 5;
  const int lane   = threadIdx.x & 31;
  const int r      = lane & 15;
  const int halfid = lane >> 4;
  const int row0   = blockIdx.x * 16;
  const int col0   = wave * 64;

  v8f acc[4] = {v8f{}, v8f{}, v8f{}, v8f{}};
  for (int e = 0; e < 512; e += 32) {
    v16h a = load_fragA(wVh + (size_t)row0 * 512 + e, 512);
#pragma unroll
    for (int j = 0; j < 4; ++j) {
      v16h bw = load_fragB(WO + (size_t)(col0 + j * 16) * 512 + e, 512);
      acc[j] = wmma16(a, bw, acc[j]);
    }
  }
#pragma unroll
  for (int j = 0; j < 4; ++j) {
    const int col = col0 + j * 16 + r;
#pragma unroll
    for (int v = 0; v < 8; ++v) {
      const int row = row0 + v + 8 * halfid;
      out[(size_t)row * 512 + col] = acc[j][v];
    }
  }
}

// ---------------------------------------------------------------------------
extern "C" void kernel_launch(void* const* d_in, const int* in_sizes, int n_in,
                              void* d_out, int out_size, void* d_ws, size_t ws_size,
                              hipStream_t stream) {
  const float* x   = (const float*)d_in[0];
  const float* W_K = (const float*)d_in[1];
  const float* W_Q = (const float*)d_in[2];
  const float* W_V = (const float*)d_in[3];
  const float* W_O = (const float*)d_in[4];
  float* out = (float*)d_out;

  const int NX = 2 * 4096 * 512;   // 4,194,304
  const int NW = 512 * 512;        //   262,144

  half_t* ws  = (half_t*)d_ws;
  half_t* xh  = ws;
  half_t* Qh  = xh  + NX;
  half_t* Kh  = Qh  + NX;
  half_t* Vth = Kh  + NX;
  half_t* wVh = Vth + NX;
  half_t* WQh = wVh + NX;
  half_t* WKh = WQh + NW;
  half_t* WVh = WKh + NW;
  half_t* WOh = WVh + NW;

  msa_cvt_f32_f16<<<(NX + 255) / 256, 256, 0, stream>>>(x,   xh,  NX);
  msa_cvt_f32_f16<<<(NW + 255) / 256, 256, 0, stream>>>(W_Q, WQh, NW);
  msa_cvt_f32_f16<<<(NW + 255) / 256, 256, 0, stream>>>(W_K, WKh, NW);
  msa_cvt_f32_f16<<<(NW + 255) / 256, 256, 0, stream>>>(W_V, WVh, NW);
  msa_cvt_f32_f16<<<(NW + 255) / 256, 256, 0, stream>>>(W_O, WOh, NW);

  msa_proj_qkv<<<dim3(512, 3), 256, 0, stream>>>(xh, WQh, WKh, WVh, Qh, Kh, Vth);
  msa_attention<<<dim3(32, 8, 2), 256, 0, stream>>>(Qh, Kh, Vth, wVh);
  msa_proj_out<<<dim3(512), 256, 0, stream>>>(wVh, WOh, out);
}
